// SelfAttentionConv_3066606649394
// MI455X (gfx1250) — matrix operand order repaired and checked by reading, not compiled, not run on hardware
//
#include <hip/hip_runtime.h>
#include <hip/hip_bf16.h>

typedef __bf16 bf16;
typedef bf16 v16bf __attribute__((ext_vector_type(16)));
typedef bf16 v8bf  __attribute__((ext_vector_type(8)));
typedef float v8f  __attribute__((ext_vector_type(8)));

#define T_LEN 2048
#define KF    128
#define HH    8
#define BB    4
#define CH    1024          // K*H
#define BH    32            // B*H
#define QK_SCALE 0.08838834764831845f   // 1/sqrt(128)

union Frag { v16bf v; v8bf g[2]; };

// ---------------------------------------------------------------------------
// Kernel 0: convert x -> bf16, repack weights transposed for B-fragments.
//   Wqr/Wkr : [j][i][c]   (5,128,1024)   from Wq/Wk (1024,128,5)
//   Wvr     : [i][c]      (128,1024)     from Wv (1024,128,1)
//   Wur     : [c][m]      (1024,128)     from Wu (128,1024)
// ---------------------------------------------------------------------------
__global__ void pack_kernel(const float* __restrict__ x,
                            const float* __restrict__ Wq, const float* __restrict__ Wk,
                            const float* __restrict__ Wv, const float* __restrict__ Wu,
                            bf16* __restrict__ xbf, bf16* __restrict__ Wqr,
                            bf16* __restrict__ Wkr, bf16* __restrict__ Wvr,
                            bf16* __restrict__ Wur) {
  int idx = blockIdx.x * blockDim.x + threadIdx.x;
  const int NX = BB * T_LEN * KF;     // 1048576
  const int NW = CH * KF * 5;         // 655360
  const int NV = CH * KF;             // 131072
  if (idx < NX) { xbf[idx] = (bf16)x[idx]; return; }
  idx -= NX;
  if (idx < NW) {
    int c = idx & 1023; int ji = idx >> 10; int i = ji & 127; int j = ji >> 7;
    Wqr[idx] = (bf16)Wq[(c * KF + i) * 5 + j];
    Wkr[idx] = (bf16)Wk[(c * KF + i) * 5 + j];
    return;
  }
  idx -= NW;
  if (idx < NV) {
    int c = idx & 1023; int i = idx >> 10;
    Wvr[idx] = (bf16)Wv[c * KF + i];
    return;
  }
  idx -= NV;
  if (idx < NV) {
    int m = idx & 127; int c = idx >> 7;
    Wur[idx] = (bf16)Wu[m * CH + c];
  }
}

// ---------------------------------------------------------------------------
// Kernel 1: fused QKV projection via WMMA GEMMs.
//   Q/K: conv ksz=5 causal => sum over 5 shifted GEMMs (T x 128) @ (128 x CH)
//   V:   ksz=1 GEMM.
//   Q stored transposed (BH,128,T) w/ bias + 1/sqrt(128); K,V row-major (BH,T,128).
//   grid: (T/16, CH/16, B*3), block: 32 (one wave per 16x16 tile)
// ---------------------------------------------------------------------------
__global__ void proj_kernel(const bf16* __restrict__ xbf,
                            const bf16* __restrict__ Wqr, const bf16* __restrict__ Wkr,
                            const bf16* __restrict__ Wvr,
                            const float* __restrict__ bq, const float* __restrict__ bk,
                            bf16* __restrict__ Qt, bf16* __restrict__ Kr,
                            bf16* __restrict__ Vr) {
  const int lane = threadIdx.x;
  const int rw = lane & 15, grp = lane >> 4, base = grp << 3;
  const int t0 = blockIdx.x << 4;
  const int c0 = blockIdx.y << 4;
  const int b = blockIdx.z / 3, which = blockIdx.z % 3;
  const bf16* xb = xbf + (size_t)b * T_LEN * KF;
  const int wrow = rw + (grp << 4);
  v8f acc = {0.f, 0.f, 0.f, 0.f, 0.f, 0.f, 0.f, 0.f};

  if (which < 2) {
    const bf16* W = which ? Wkr : Wqr;
#pragma unroll
    for (int j = 0; j < 5; ++j) {
      const int tr = t0 + rw - 4 + j;        // causal left-pad: rows < 0 are zero
      const bf16* xr = xb + (size_t)tr * KF;
#pragma unroll
      for (int kc = 0; kc < 4; ++kc) {
        Frag a;
        if (tr >= 0) {
          a.g[0] = *(const v8bf*)(xr + kc * 32 + base);
          a.g[1] = *(const v8bf*)(xr + kc * 32 + 16 + base);
        } else {
#pragma unroll
          for (int i = 0; i < 16; ++i) a.v[i] = (bf16)0.0f;
        }
        Frag w;
        const bf16* wp = W + ((size_t)(j * KF + kc * 32 + base) * CH) + c0 + rw;
#pragma unroll
        for (int i = 0; i < 8; ++i) { w.v[i] = wp[(size_t)i * CH]; w.v[8 + i] = wp[(size_t)(16 + i) * CH]; }
        acc = __builtin_amdgcn_wmma_f32_16x16x32_bf16(false, a.v, false, w.v,
                                                      (short)0, acc, false, false);
      }
    }
    const int c = c0 + rw, h = c & 7, kf = c >> 3;
    const int bh = (b << 3) | h;
    if (which == 0) {
      const float bias = bq[c];
      v8bf pk;
#pragma unroll
      for (int v = 0; v < 8; ++v) pk[v] = (bf16)((acc[v] + bias) * QK_SCALE);
      *(v8bf*)(Qt + ((size_t)bh * KF + kf) * T_LEN + t0 + base) = pk;   // transposed
    } else {
      const float bias = bk[c];
#pragma unroll
      for (int v = 0; v < 8; ++v)
        Kr[((size_t)bh * T_LEN + t0 + base + v) * KF + kf] = (bf16)(acc[v] + bias);
    }
  } else {
    const bf16* xr = xb + (size_t)(t0 + rw) * KF;
#pragma unroll
    for (int kc = 0; kc < 4; ++kc) {
      Frag a, w;
      a.g[0] = *(const v8bf*)(xr + kc * 32 + base);
      a.g[1] = *(const v8bf*)(xr + kc * 32 + 16 + base);
      const bf16* wp = Wvr + ((size_t)(kc * 32 + base) * CH) + c0 + rw;
#pragma unroll
      for (int i = 0; i < 8; ++i) { w.v[i] = wp[(size_t)i * CH]; w.v[8 + i] = wp[(size_t)(16 + i) * CH]; }
      acc = __builtin_amdgcn_wmma_f32_16x16x32_bf16(false, a.v, false, w.v,
                                                    (short)0, acc, false, false);
    }
    const int c = c0 + rw, h = c & 7, kf = c >> 3;
    const int bh = (b << 3) | h;
#pragma unroll
    for (int v = 0; v < 8; ++v)
      Vr[((size_t)bh * T_LEN + t0 + base + v) * KF + kf] = (bf16)acc[v];
  }
}

// ---------------------------------------------------------------------------
// Kernel 2: causal flash attention, one wave per (head, 16-query tile).
//   S^T = K_tile @ Q^T  (C-fragment of S^T == A-fragment of P, zero shuffles)
//   O  += P @ V         (V row-major is the B-fragment directly)
//   Output written h-major into Obf (B,T,1024) bf16 for the final projection.
//   grid: (T/16, 32), block: 32
// ---------------------------------------------------------------------------
__global__ void attn_kernel(const bf16* __restrict__ Qt, const bf16* __restrict__ Kr,
                            const bf16* __restrict__ Vr, bf16* __restrict__ Obf) {
  const int lane = threadIdx.x;
  const int rw = lane & 15, grp = lane >> 4, base = grp << 3;
  const int qt = blockIdx.x;
  const int bh = blockIdx.y;
  const int b = bh >> 3, h = bh & 7;

  Frag qf[4];                                  // loop-invariant Q^T B-fragments
#pragma unroll
  for (int kc = 0; kc < 4; ++kc) {
    const bf16* p = Qt + ((size_t)bh * KF + kc * 32 + base) * T_LEN + (qt << 4) + rw;
#pragma unroll
    for (int i = 0; i < 8; ++i) { qf[kc].v[i] = p[(size_t)i * T_LEN]; qf[kc].v[8 + i] = p[(size_t)(16 + i) * T_LEN]; }
  }

  float m_st = -1e30f, l_st = 0.0f;            // per-lane = per-query (rw) stats
  v8f o[8];
#pragma unroll
  for (int ft = 0; ft < 8; ++ft) o[ft] = (v8f){0.f, 0.f, 0.f, 0.f, 0.f, 0.f, 0.f, 0.f};

  const int q = (qt << 4) + rw;
  const int kbmax = ((qt << 4) + 15) >> 5;
  const bf16* Kb = Kr + (size_t)bh * T_LEN * KF;
  const bf16* Vb = Vr + (size_t)bh * T_LEN * KF;

  for (int kb = 0; kb <= kbmax; ++kb) {
    const int k0 = kb << 5;
    v8f st0 = {0.f, 0.f, 0.f, 0.f, 0.f, 0.f, 0.f, 0.f};
    v8f st1 = {0.f, 0.f, 0.f, 0.f, 0.f, 0.f, 0.f, 0.f};
#pragma unroll
    for (int kc = 0; kc < 4; ++kc) {
      Frag a0, a1;
      const bf16* p0 = Kb + (size_t)(k0 + rw) * KF + kc * 32;
      const bf16* p1 = Kb + (size_t)(k0 + 16 + rw) * KF + kc * 32;
      a0.g[0] = *(const v8bf*)(p0 + base);
      a0.g[1] = *(const v8bf*)(p0 + 16 + base);
      a1.g[0] = *(const v8bf*)(p1 + base);
      a1.g[1] = *(const v8bf*)(p1 + 16 + base);
      st0 = __builtin_amdgcn_wmma_f32_16x16x32_bf16(false, a0.v, false, qf[kc].v,
                                                    (short)0, st0, false, false);
      st1 = __builtin_amdgcn_wmma_f32_16x16x32_bf16(false, a1.v, false, qf[kc].v,
                                                    (short)0, st1, false, false);
    }
    // causal mask: key index = k0 (+16) + base + v; query = q (this lane's column)
    float mx = -1e30f;
#pragma unroll
    for (int v = 0; v < 8; ++v) {
      if (k0 + base + v > q)      st0[v] = -1e30f;
      if (k0 + 16 + base + v > q) st1[v] = -1e30f;
      mx = fmaxf(mx, fmaxf(st0[v], st1[v]));
    }
    mx = fmaxf(mx, __shfl_xor(mx, 16, 32));
    const float mn = fmaxf(m_st, mx);
    const float corr = __expf(m_st - mn);
    m_st = mn;

    Frag p;                                    // P as A-fragment, built in-lane
    float s = 0.0f;
#pragma unroll
    for (int v = 0; v < 8; ++v) {
      const float e0 = __expf(st0[v] - mn);
      const float e1 = __expf(st1[v] - mn);
      p.v[v]     = (bf16)e0;
      p.v[8 + v] = (bf16)e1;
      s += e0 + e1;
    }
    s += __shfl_xor(s, 16, 32);
    l_st = l_st * corr + s;

    float cf[8];                               // rescale factors for O's M rows
#pragma unroll
    for (int v = 0; v < 8; ++v) cf[v] = __shfl(corr, base + v, 32);
#pragma unroll
    for (int ft = 0; ft < 8; ++ft)
#pragma unroll
      for (int v = 0; v < 8; ++v) o[ft][v] *= cf[v];

#pragma unroll
    for (int ft = 0; ft < 8; ++ft) {
      Frag vf;
      const bf16* vp = Vb + (size_t)(k0 + base) * KF + (ft << 4) + rw;
#pragma unroll
      for (int i = 0; i < 8; ++i) { vf.v[i] = vp[(size_t)i * KF]; vf.v[8 + i] = vp[(size_t)(16 + i) * KF]; }
      o[ft] = __builtin_amdgcn_wmma_f32_16x16x32_bf16(false, p.v, false, vf.v,
                                                      (short)0, o[ft], false, false);
    }
  }

  const float inv = 1.0f / l_st;
  float iv[8];
#pragma unroll
  for (int v = 0; v < 8; ++v) iv[v] = __shfl(inv, base + v, 32);
  bf16* ob = Obf + ((size_t)b * T_LEN + (qt << 4)) * CH + (h << 7);
#pragma unroll
  for (int ft = 0; ft < 8; ++ft)
#pragma unroll
    for (int v = 0; v < 8; ++v)
      ob[(size_t)(base + v) * CH + (ft << 4) + rw] = (bf16)(o[ft][v] * iv[v]);
}

// ---------------------------------------------------------------------------
// Kernel 3: final projection  out = Obf(8192x1024) @ Wu^T(1024x128) + bu, fp32 out
//   grid: (B*T/16, 128/16), block: 32
// ---------------------------------------------------------------------------
__global__ void outproj_kernel(const bf16* __restrict__ Obf, const bf16* __restrict__ Wur,
                               const float* __restrict__ bu, float* __restrict__ out) {
  const int lane = threadIdx.x;
  const int rw = lane & 15, grp = lane >> 4, base = grp << 3;
  const int r0 = blockIdx.x << 4;
  const int n0 = blockIdx.y << 4;
  v8f acc = {0.f, 0.f, 0.f, 0.f, 0.f, 0.f, 0.f, 0.f};
  const bf16* ar = Obf + (size_t)(r0 + rw) * CH;
#pragma unroll 4
  for (int kc = 0; kc < 32; ++kc) {
    Frag a, w;
    a.g[0] = *(const v8bf*)(ar + kc * 32 + base);
    a.g[1] = *(const v8bf*)(ar + kc * 32 + 16 + base);
    const bf16* wp = Wur + (size_t)(kc * 32 + base) * KF + n0 + rw;
#pragma unroll
    for (int i = 0; i < 8; ++i) { w.v[i] = wp[(size_t)i * KF]; w.v[8 + i] = wp[(size_t)(16 + i) * KF]; }
    acc = __builtin_amdgcn_wmma_f32_16x16x32_bf16(false, a.v, false, w.v,
                                                  (short)0, acc, false, false);
  }
  const int col = n0 + rw;
  const float bias = bu[col];
#pragma unroll
  for (int v = 0; v < 8; ++v)
    out[(size_t)(r0 + base + v) * KF + col] = acc[v] + bias;
}

// ---------------------------------------------------------------------------
extern "C" void kernel_launch(void* const* d_in, const int* in_sizes, int n_in,
                              void* d_out, int out_size, void* d_ws, size_t ws_size,
                              hipStream_t stream) {
  const float* x  = (const float*)d_in[0];
  const float* Wq = (const float*)d_in[1];
  const float* bq = (const float*)d_in[2];
  const float* Wk = (const float*)d_in[3];
  const float* bk = (const float*)d_in[4];
  const float* Wv = (const float*)d_in[5];
  const float* Wu = (const float*)d_in[6];
  const float* bu = (const float*)d_in[7];
  float* out = (float*)d_out;

  char* ws = (char*)d_ws;
  size_t off = 0;
  auto take = [&](size_t elems) -> bf16* {
    bf16* p = (bf16*)(ws + off);
    off = (off + elems * sizeof(bf16) + 255) & ~(size_t)255;
    return p;
  };
  bf16* xbf = take((size_t)BB * T_LEN * KF);      //  2 MB
  bf16* Wqr = take((size_t)5 * KF * CH);          //  1.25 MB
  bf16* Wkr = take((size_t)5 * KF * CH);          //  1.25 MB
  bf16* Wvr = take((size_t)KF * CH);              //  0.25 MB
  bf16* Wur = take((size_t)CH * KF);              //  0.25 MB
  bf16* Qt  = take((size_t)BH * KF * T_LEN);      // 16 MB (transposed)
  bf16* Kr  = take((size_t)BH * T_LEN * KF);      // 16 MB
  bf16* Vr  = take((size_t)BH * T_LEN * KF);      // 16 MB
  bf16* Obf = take((size_t)BB * T_LEN * CH);      // 16 MB

  const int NPACK = BB * T_LEN * KF + 5 * KF * CH + KF * CH + CH * KF;
  pack_kernel<<<dim3((NPACK + 255) / 256), dim3(256), 0, stream>>>(
      x, Wq, Wk, Wv, Wu, xbf, Wqr, Wkr, Wvr, Wur);

  proj_kernel<<<dim3(T_LEN / 16, CH / 16, BB * 3), dim3(32), 0, stream>>>(
      xbf, Wqr, Wkr, Wvr, bq, bk, Qt, Kr, Vr);

  attn_kernel<<<dim3(T_LEN / 16, BH), dim3(32), 0, stream>>>(Qt, Kr, Vr, Obf);

  outproj_kernel<<<dim3(BB * T_LEN / 16, KF / 16), dim3(32), 0, stream>>>(
      Obf, Wur, bu, out);
}
